// MultiHeadSelfAttention_50783693308250
// MI455X (gfx1250) — matrix operand order repaired and checked by reading, not compiled, not run on hardware
//
#include <hip/hip_runtime.h>

// ---------------------------------------------------------------------------
// MultiHeadSelfAttention forward for MI455X (gfx1250, wave32, WMMA + TDM).
//
// B=2, S=2048, E=768, H=12, D=64.  d_out = out[2,2048,768] ++ attn[2,12,2048,2048]
//
// Roofline: ~45 GFLOP total; dominant traffic is the 402 MB fp32 attn-prob
// write (~17 us at 23.3 TB/s).  Scores for each 16-query strip stay resident
// in LDS (128 KB f32 + 64 KB bf16 + 32 KB K-stage << 320 KB/WGP), softmax and
// the PV GEMM are fused, probs are written exactly once with nontemporal
// stores.  All GEMMs use v_wmma_f32_16x16x32_bf16 (bf16 in, f32 accumulate).
// K tiles are DMA'd into LDS by the Tensor Data Mover (tensor_load_to_lds,
// TENSORcnt, ping-pong double buffer) so score-phase B fragments come from
// ds_load_b128 instead of per-lane strided VMEM.
// ---------------------------------------------------------------------------

#define E_DIM 768
#define N_HEADS 12
#define HD 64
#define SEQ 2048
#define BATCH 2
#define MROWS (BATCH * SEQ) // 4096

typedef __bf16 bf16_t;
typedef __attribute__((ext_vector_type(16))) bf16_t v16b;
typedef __attribute__((ext_vector_type(8)))  float  v8f;
typedef unsigned int uv4 __attribute__((ext_vector_type(4)));
typedef unsigned int u32x4 __attribute__((ext_vector_type(4)));
typedef int i32x4v __attribute__((ext_vector_type(4)));
typedef int i32x8v __attribute__((ext_vector_type(8)));

union Frag { uv4 q[2]; v16b v; };

__device__ __forceinline__ bf16_t f2bf(float f) {
  unsigned int u = __builtin_bit_cast(unsigned int, f);
  u += 0x7FFFu + ((u >> 16) & 1u);              // round-to-nearest-even
  unsigned short h = (unsigned short)(u >> 16);
  return __builtin_bit_cast(bf16_t, h);
}

// A-matrix 16x32 bf16 fragment: lane&15 = M row;
// lanes 0-15 hold K 0..15, lanes 16-31 hold K 16..31.
__device__ __forceinline__ v16b load_a_frag(const bf16_t* base, int ld, int lane) {
  int r    = lane & 15;
  int koff = (lane >> 4) ? 16 : 0;
  const bf16_t* p = base + (size_t)r * ld + koff;
  Frag f;
  f.q[0] = *reinterpret_cast<const uv4*>(p);        // K koff..koff+7
  f.q[1] = *reinterpret_cast<const uv4*>(p + 8);    // K koff+8..koff+15
  return f.v;
}

// B-matrix 32x16 bf16 fragment where B(k,n) = base[n*ld + k]
// (operand stored N-major / "transposed", contiguous in K).
// lane&15 = N col; lanes 0-15 hold K 0..15, lanes 16-31 hold K 16..31.
__device__ __forceinline__ v16b load_bt_frag(const bf16_t* base, int ld, int lane) {
  int n    = lane & 15;
  int koff = (lane >> 4) ? 16 : 0;
  const bf16_t* p = base + (size_t)n * ld + koff;
  Frag f;
  f.q[0] = *reinterpret_cast<const uv4*>(p);        // K koff..koff+7
  f.q[1] = *reinterpret_cast<const uv4*>(p + 8);    // K koff+8..koff+15
  return f.v;
}

__device__ __forceinline__ v8f wmma_bf16(v16b a, v16b b, v8f c) {
  return __builtin_amdgcn_wmma_f32_16x16x32_bf16(
      /*neg_a=*/false, a, /*neg_b=*/false, b,
      /*c_mod=*/(short)0, c, /*reuse_a=*/false, /*reuse_b=*/false);
}

// Tensor Data Mover: DMA a rows x cols 2-byte-element 2-D tile from global
// memory into LDS.  D# bitfields per CDNA5 ISA.
__device__ __forceinline__ void tdm_load_2d_bf16(unsigned lds_byte_addr,
                                                 const bf16_t* gsrc,
                                                 unsigned rows, unsigned cols,
                                                 unsigned row_stride_elems) {
  unsigned long long ga = (unsigned long long)(__UINTPTR_TYPE__)gsrc;
  u32x4 g0;
  g0[0] = 1u;                                            // count=1 user D#
  g0[1] = lds_byte_addr;                                 // lds_addr [63:32]
  g0[2] = (unsigned)(ga & 0xFFFFFFFFu);                  // global_addr lo
  g0[3] = (unsigned)((ga >> 32) & 0x01FFFFFFu) | (2u << 30); // addr hi | type=2
  i32x8v g1;
  g1[0] = (int)(1u << 16);                               // data_size=1 (2 B)
  g1[1] = (int)((cols & 0xFFFFu) << 16);                 // tensor_dim0 lo16
  g1[2] = (int)((cols >> 16) | ((rows & 0xFFFFu) << 16));// dim0 hi / dim1 lo
  g1[3] = (int)((rows >> 16) | ((cols & 0xFFFFu) << 16));// dim1 hi / tile_dim0
  g1[4] = (int)(rows & 0xFFFFu);                         // tile_dim1 (dim2=0)
  g1[5] = (int)row_stride_elems;                         // dim0_stride lo32
  g1[6] = 0;
  g1[7] = 0;
  i32x4v z4 = {};
#if defined(__clang_major__) && __clang_major__ >= 23
  i32x8v z8 = {};
  __builtin_amdgcn_tensor_load_to_lds(g0, g1, z4, z4, z8, 0);
#else
  __builtin_amdgcn_tensor_load_to_lds(g0, g1, z4, z4, 0);
#endif
}

// ---------------------------------------------------------------------------
__global__ void cvt_f32_to_bf16(const float* __restrict__ src,
                                bf16_t* __restrict__ dst, int n) {
  int i = blockIdx.x * blockDim.x + threadIdx.x;
  if (i < n) dst[i] = f2bf(src[i]);
}

// ---------------------------------------------------------------------------
// y[m,n] = sum_k X[m,k] * W[n,k] + bias[n]     (X: MROWSxE bf16, W: ExE bf16)
// mode 0: Q  -> dstb[b][h][s][d] bf16, scaled by 1/sqrt(D)
// mode 1: K  -> dstb[b][h][s][d] bf16
// mode 2: V  -> dstb[b][h][d][s] bf16  (transposed per head for PV GEMM)
// mode 3: O  -> dstf[m*E + n] f32      (final output projection)
// Block: 256 threads = 8 waves, tile 32(M) x 64(N); each wave one 16x16 tile.
__global__ void proj_gemm(const bf16_t* __restrict__ X,
                          const bf16_t* __restrict__ W,
                          const float* __restrict__ bias,
                          bf16_t* __restrict__ dstb,
                          float* __restrict__ dstf,
                          int mode) {
  int w = threadIdx.x >> 5, lane = threadIdx.x & 31;
  int mt = blockIdx.x * 32 + (w >> 2) * 16;
  int nt = blockIdx.y * 64 + (w & 3) * 16;

  const bf16_t* abase = X + (size_t)mt * E_DIM;
  const bf16_t* bbase = W + (size_t)nt * E_DIM;

  v8f acc = {};
#pragma unroll 4
  for (int k0 = 0; k0 < E_DIM; k0 += 32) {
    v16b a = load_a_frag(abase + k0, E_DIM, lane);
    v16b b = load_bt_frag(bbase + k0, E_DIM, lane);
    acc = wmma_bf16(a, b, acc);
  }

  int n   = nt + (lane & 15);
  float bs = bias[n];
  int rb  = (lane >> 4) ? 8 : 0;

  if (mode == 3) {
#pragma unroll
    for (int r = 0; r < 8; ++r)
      dstf[(size_t)(mt + rb + r) * E_DIM + n] = acc[r] + bs;
  } else {
    int h = n >> 6, d = n & 63;
#pragma unroll
    for (int r = 0; r < 8; ++r) {
      int m = mt + rb + r;
      int b_ = m >> 11, s = m & (SEQ - 1);
      float val = acc[r] + bs;
      if (mode == 0) val *= 0.125f; // 1/sqrt(64) folded into Q
      size_t off;
      if (mode == 2)
        off = (((size_t)(b_ * N_HEADS + h)) * HD + d) * SEQ + s; // V^T
      else
        off = (((size_t)(b_ * N_HEADS + h)) * SEQ + s) * HD + d; // Q,K
      dstb[off] = f2bf(val);
    }
  }
}

// ---------------------------------------------------------------------------
// Fused attention: one block per (b*h, 16-query strip).
// LDS: sc[16][2048] f32 (128K) | pb[16][2048] bf16 (64K) | rmax/rsum | K stage 2x16K
#define SC_BYTES   (16 * SEQ * 4)
#define PB_BYTES   (16 * SEQ * 2)
#define STAT_BYTES 128
#define STG_OFF    (SC_BYTES + PB_BYTES + STAT_BYTES)
#define STG_CHUNK  128                       // K rows per TDM chunk
#define SMEM_TOTAL (STG_OFF + 2 * STG_CHUNK * HD * 2)

__global__ void attention_fused(const bf16_t* __restrict__ qmat,
                                const bf16_t* __restrict__ kmat,
                                const bf16_t* __restrict__ vt,
                                bf16_t* __restrict__ ctx,
                                float* __restrict__ attn) {
  extern __shared__ char smem_raw[];
  float*  sc   = reinterpret_cast<float*>(smem_raw);
  bf16_t* pb   = reinterpret_cast<bf16_t*>(smem_raw + SC_BYTES);
  float*  rmax = reinterpret_cast<float*>(smem_raw + SC_BYTES + PB_BYTES);
  float*  rsum = rmax + 16;
  bf16_t* kstage = reinterpret_cast<bf16_t*>(smem_raw + STG_OFF);

  int bh = blockIdx.y;
  int q0 = blockIdx.x * 16;
  int w = threadIdx.x >> 5, lane = threadIdx.x & 31;

  // ---- Phase 1: scores S = Q K^T (Q pre-scaled by 1/sqrt(D)) -> LDS ----
  // K head panel is DMA'd chunk-by-chunk into an LDS ping-pong buffer by the
  // Tensor Data Mover; wave 0 drives the DMA, TENSORcnt orders it.
  const bf16_t* khead = kmat + (size_t)bh * SEQ * HD;
  const unsigned stg_lds = (unsigned)(__UINTPTR_TYPE__)(void*)kstage;

  if (w == 0)
    tdm_load_2d_bf16(stg_lds, khead, STG_CHUNK, HD, HD);

  const bf16_t* qbase = qmat + ((size_t)bh * SEQ + q0) * HD;
  v16b aq0 = load_a_frag(qbase + 0,  HD, lane);
  v16b aq1 = load_a_frag(qbase + 32, HD, lane);

  int rb  = (lane >> 4) ? 8 : 0;
  int col = lane & 15;
  for (int t = 0; t < SEQ / STG_CHUNK; ++t) {
    if (w == 0) {
      if (t + 1 < SEQ / STG_CHUNK) {
        tdm_load_2d_bf16(stg_lds + (unsigned)(((t + 1) & 1) * STG_CHUNK * HD * 2),
                         khead + (size_t)(t + 1) * STG_CHUNK * HD,
                         STG_CHUNK, HD, HD);
        __builtin_amdgcn_s_wait_tensorcnt(1);  // chunk t complete (in-order)
      } else {
        __builtin_amdgcn_s_wait_tensorcnt(0);
      }
    }
    __syncthreads();                           // chunk t visible to all waves

    const bf16_t* kb = kstage + (size_t)((t & 1) * STG_CHUNK + w * 16) * HD;
    v8f c = {};
    c = wmma_bf16(aq0, load_bt_frag(kb + 0,  HD, lane), c);
    c = wmma_bf16(aq1, load_bt_frag(kb + 32, HD, lane), c);
    int n0 = t * STG_CHUNK + w * 16;
#pragma unroll
    for (int r = 0; r < 8; ++r) sc[(size_t)(rb + r) * SEQ + n0 + col] = c[r];
    __syncthreads();                           // reads done before buf reuse
  }

  // ---- Phase 2: per-row max & expsum (one wave per row, 2 rows/wave) ----
#pragma unroll
  for (int rr = 0; rr < 2; ++rr) {
    int row = w + rr * 8;
    const float* srow = sc + (size_t)row * SEQ;
    float m = -3.0e38f;
    for (int i = lane; i < SEQ; i += 32) m = fmaxf(m, srow[i]);
    for (int off = 16; off > 0; off >>= 1) m = fmaxf(m, __shfl_xor(m, off, 32));
    float s = 0.f;
    for (int i = lane; i < SEQ; i += 32) s += __expf(srow[i] - m);
    for (int off = 16; off > 0; off >>= 1) s += __shfl_xor(s, off, 32);
    if (lane == 0) { rmax[row] = m; rsum[row] = 1.0f / s; }
  }
  __syncthreads();

  // ---- Phase 3: probs -> global (nontemporal, write-once) + bf16 LDS ----
  float* attn_base = attn + ((size_t)bh * SEQ + q0) * SEQ;
  for (int i = threadIdx.x; i < 16 * SEQ; i += 256) {
    int row = i >> 11;
    float p = __expf(sc[i] - rmax[row]) * rsum[row];
    __builtin_nontemporal_store(p, attn_base + i);
    pb[i] = f2bf(p);
  }
  __syncthreads();

  // ---- Phase 4: O = P V via WMMA, split-K across wave pairs ----
  int ctile = w & 3, khalf = w >> 2;   // 4 column tiles x 2 K-halves
  const bf16_t* vbase = vt + ((size_t)bh * HD + ctile * 16) * SEQ;
  v8f acc = {};
  int kbeg = khalf * (SEQ / 2);
#pragma unroll 4
  for (int kk = kbeg; kk < kbeg + SEQ / 2; kk += 32) {
    v16b ap = load_a_frag(pb + kk, SEQ, lane);        // ds_load_b128 from LDS
    v16b bv = load_bt_frag(vbase + kk, SEQ, lane);    // V^T rows, contiguous
    acc = wmma_bf16(ap, bv, acc);
  }

  float* scratch = sc;                 // score strip is dead now; reuse
  if (w >= 4) {
#pragma unroll
    for (int r = 0; r < 8; ++r) scratch[(w - 4) * 256 + r * 32 + lane] = acc[r];
  }
  __syncthreads();
  if (w < 4) {
#pragma unroll
    for (int r = 0; r < 8; ++r) acc[r] += scratch[w * 256 + r * 32 + lane];
    int b_ = bh / N_HEADS, h = bh % N_HEADS;
    int d = ctile * 16 + (lane & 15);
    bf16_t* cbase = ctx + ((size_t)b_ * SEQ + q0) * E_DIM + h * HD + d;
#pragma unroll
    for (int r = 0; r < 8; ++r) cbase[(size_t)(rb + r) * E_DIM] = f2bf(acc[r]);
  }
}

// ---------------------------------------------------------------------------
extern "C" void kernel_launch(void* const* d_in, const int* in_sizes, int n_in,
                              void* d_out, int out_size, void* d_ws, size_t ws_size,
                              hipStream_t stream) {
  const float* x  = (const float*)d_in[0];
  const float* Wq = (const float*)d_in[1];
  const float* bq = (const float*)d_in[2];
  const float* Wk = (const float*)d_in[3];
  const float* bk = (const float*)d_in[4];
  const float* Wv = (const float*)d_in[5];
  const float* bv = (const float*)d_in[6];
  const float* Wo = (const float*)d_in[7];
  const float* bo = (const float*)d_in[8];

  float* out  = (float*)d_out;
  float* attn = out + (size_t)MROWS * E_DIM; // out first, then attn probs

  // Workspace layout (bf16 elements); total ~36 MB.
  const size_t XB_N  = (size_t)MROWS * E_DIM;              // 3,145,728
  const size_t W_N   = (size_t)E_DIM * E_DIM;              //   589,824
  const size_t QKV_N = (size_t)BATCH * N_HEADS * SEQ * HD; // 3,145,728
  bf16_t* xb  = (bf16_t*)d_ws;
  bf16_t* wqb = xb  + XB_N;
  bf16_t* wkb = wqb + W_N;
  bf16_t* wvb = wkb + W_N;
  bf16_t* wob = wvb + W_N;
  bf16_t* qm  = wob + W_N;
  bf16_t* km  = qm  + QKV_N;
  bf16_t* vtm = km  + QKV_N;
  bf16_t* ctx = vtm + QKV_N;

  // 1) fp32 -> bf16 conversions
  cvt_f32_to_bf16<<<(int)((XB_N + 255) / 256), 256, 0, stream>>>(x,  xb,  (int)XB_N);
  cvt_f32_to_bf16<<<(int)((W_N  + 255) / 256), 256, 0, stream>>>(Wq, wqb, (int)W_N);
  cvt_f32_to_bf16<<<(int)((W_N  + 255) / 256), 256, 0, stream>>>(Wk, wkb, (int)W_N);
  cvt_f32_to_bf16<<<(int)((W_N  + 255) / 256), 256, 0, stream>>>(Wv, wvb, (int)W_N);
  cvt_f32_to_bf16<<<(int)((W_N  + 255) / 256), 256, 0, stream>>>(Wo, wob, (int)W_N);

  // 2) Q/K/V projections (WMMA)
  dim3 gproj(MROWS / 32, E_DIM / 64);
  proj_gemm<<<gproj, 256, 0, stream>>>(xb, wqb, bq, qm,  nullptr, 0);
  proj_gemm<<<gproj, 256, 0, stream>>>(xb, wkb, bk, km,  nullptr, 1);
  proj_gemm<<<gproj, 256, 0, stream>>>(xb, wvb, bv, vtm, nullptr, 2);

  // 3) fused scores + softmax + PV (scores LDS-resident, TDM-fed K tiles)
  dim3 gattn(SEQ / 16, BATCH * N_HEADS);
  attention_fused<<<gattn, 256, SMEM_TOTAL, stream>>>(qm, km, vtm, ctx, attn);

  // 4) output projection -> fp32 out
  proj_gemm<<<gproj, 256, 0, stream>>>(ctx, wob, bo, nullptr, out, 3);
}
